// RelaxedOTLoss_59339268162035
// MI455X (gfx1250) — matrix operand order repaired and checked by reading, not compiled, not run on hardware
//
#include <hip/hip_runtime.h>

typedef _Float16 half_t;
typedef __attribute__((ext_vector_type(16))) _Float16 v16h;
typedef __attribute__((ext_vector_type(8)))  _Float16 v8h;
typedef __attribute__((ext_vector_type(8)))  float    v8f;
typedef __attribute__((ext_vector_type(4)))  int      v4i;

union HFrag { v16h v; v8h h[2]; };

// gfx1250 async global->LDS path (ASYNCcnt), guarded so a builtin name miss
// falls back to plain b128 copies instead of breaking the build.
#if defined(__has_builtin)
# if __has_builtin(__builtin_amdgcn_global_load_async_to_lds_b128) && \
     __has_builtin(__builtin_amdgcn_s_wait_asynccnt)
#  define HAVE_ASYNC_LDS 1
# endif
#endif
#ifndef HAVE_ASYNC_LDS
# define HAVE_ASYNC_LDS 0
#endif

__device__ __forceinline__ unsigned fenc(float f) {
    unsigned u = __float_as_uint(f);
    return (u & 0x80000000u) ? ~u : (u | 0x80000000u);
}
__device__ __forceinline__ float fdec(unsigned e) {
    return (e & 0x80000000u) ? __uint_as_float(e & 0x7FFFFFFFu)
                             : __uint_as_float(~e);
}

// ---------------------------------------------------------------------------
// One-time weight pre-pack: fp32 [Cout][Cin][3][3] -> f16 in exact WMMA
// B-fragment order [ct][cc][tap][n][k32], K zero-padded to a multiple of 32.
// ---------------------------------------------------------------------------
__global__ void pack_weights(const float* __restrict__ wgt, half_t* __restrict__ wpk,
                             int Cin, int nch, int total)
{
    int idx = blockIdx.x * blockDim.x + threadIdx.x;
    if (idx >= total) return;
    int k  = idx & 31;
    int n  = (idx >> 5) & 15;
    int tp = (idx >> 9) % 9;
    int t2 = (idx >> 9) / 9;      // ct*nch + cc
    int cc = t2 % nch;
    int ct = t2 / nch;
    int ci = cc * 32 + k;
    int co = ct * 16 + n;
    float v = (ci < Cin) ? wgt[((size_t)co * Cin + ci) * 9 + tp] : 0.f;
    wpk[idx] = (half_t)v;
}

// ---------------------------------------------------------------------------
// conv3x3 (pad 1) + bias + relu, implicit GEMM on v_wmma_f32_16x16x32_f16.
// NHWC f16 activations: each pixel's K-slice is contiguous, so the LDS strip
// is staged with 16B b128 copies (async global->LDS when available).
// MT = 1 or 2 pixel tiles of 16: each B fragment (from pre-packed global
// weights, L2-resident) feeds MT WMMAs. Epilogue transposes through LDS so
// stores are 16B contiguous in NHWC.
// ---------------------------------------------------------------------------
template <int MT>
__global__ __launch_bounds__(32) void conv3x3_wmma(
    const half_t* __restrict__ inH,   // NHWC f16 activations (null on layer 0)
    const float*  __restrict__ inF,   // NCHW fp32 images (layer 0, normalized)
    const half_t* __restrict__ wpk,   // packed f16 weights (B-frag order)
    const float*  __restrict__ bias,
    half_t* __restrict__ out,         // NHWC f16
    int Cin, int Cout, int H, int W)
{
    __shared__ __attribute__((aligned(16))) half_t sIn[3 * 34 * 32];
    __shared__ __attribute__((aligned(16))) half_t sOut[2 * 16 * 16];

    const int lane = threadIdx.x;
    const int x0 = blockIdx.x * 16 * MT;
    const int y0 = blockIdx.y;
    const int ctiles = Cout >> 4;
    const int b  = blockIdx.z / ctiles;
    const int ct = blockIdx.z % ctiles;
    const int nch = (Cin + 31) >> 5;
    const int pxs = 16 * MT + 2;

    const float nmean[3] = {0.485f, 0.456f, 0.406f};
    const float nistd[3] = {1.0f / 0.229f, 1.0f / 0.224f, 1.0f / 0.225f};

    const v8f zero8 = {0.f, 0.f, 0.f, 0.f, 0.f, 0.f, 0.f, 0.f};
    v8f acc[MT];
    #pragma unroll
    for (int t = 0; t < MT; ++t) acc[t] = zero8;

    const int mn  = lane & 15;
    const int hbA = (lane >> 4) * 8;
    const int hbB = (lane >> 4) * 8;

    for (int cc = 0; cc < nch; ++cc) {
        const int c0 = cc << 5;
        if (inF) {
            // layer 0: NCHW fp32 gather + normalize (Cin=3, one chunk)
            for (int i = lane; i < 3 * pxs * 32; i += 32) {
                int ch = i & 31;
                int px = (i >> 5) % pxs;
                int r  = (i >> 5) / pxs;
                int y = y0 + r - 1, x = x0 + px - 1, c = c0 + ch;
                float v = 0.f;
                if (c < 3 && (unsigned)y < (unsigned)H && (unsigned)x < (unsigned)W)
                    v = (inF[(((size_t)b * 3 + c) * H + y) * W + x] - nmean[c]) * nistd[c];
                sIn[(r * 34 + px) * 32 + ch] = (half_t)v;
            }
            __syncthreads();
        } else {
            // NHWC f16: 16B chunks, async copy global -> LDS when available
            const int chunks = 3 * pxs * 4;
            for (int i = lane; i < chunks; i += 32) {
                int ch8 = (i & 3) * 8;
                int px  = (i >> 2) % pxs;
                int r   = (i >> 2) / pxs;
                int y = y0 + r - 1, x = x0 + px - 1;
                int lidx = (r * 34 + px) * 32 + ch8;
                if ((unsigned)y < (unsigned)H && (unsigned)x < (unsigned)W) {
                    const half_t* g = inH + (((size_t)b * H + y) * W + x) * Cin + c0 + ch8;
#if HAVE_ASYNC_LDS
                    __builtin_amdgcn_global_load_async_to_lds_b128(
                        (__attribute__((address_space(1))) v4i*)(v4i*)g,
                        (__attribute__((address_space(3))) v4i*)(v4i*)&sIn[lidx], 0, 0);
#else
                    *(v8h*)&sIn[lidx] = *(const v8h*)g;
#endif
                } else {
                    v8h z;
                    #pragma unroll
                    for (int q = 0; q < 8; ++q) z[q] = (half_t)0.f;
                    *(v8h*)&sIn[lidx] = z;
                }
            }
#if HAVE_ASYNC_LDS
            __builtin_amdgcn_s_wait_asynccnt(0);
#endif
            __syncthreads();
        }

        const half_t* wl = wpk + ((size_t)(ct * nch + cc) * 9) * 16 * 32;
        #pragma unroll
        for (int tp = 0; tp < 9; ++tp) {
            const int r = tp / 3, s = tp % 3;
            HFrag bm;
            const half_t* pb = wl + (tp * 16 + mn) * 32 + hbB;
            bm.h[0] = *(const v8h*)pb;
            bm.h[1] = *(const v8h*)(pb + 16);
            #pragma unroll
            for (int t = 0; t < MT; ++t) {
                HFrag a;
                const half_t* pa = &sIn[(r * 34 + 16 * t + mn + s) * 32 + hbA];
                a.h[0] = *(const v8h*)pa;
                a.h[1] = *(const v8h*)(pa + 16);
                acc[t] = __builtin_amdgcn_wmma_f32_16x16x32_f16(
                    false, a.v, false, bm.v, (short)0, acc[t], false, false);
            }
        }
        __syncthreads();
    }

    // epilogue: bias + relu, transpose through LDS -> 16B NHWC stores
    const int co = ct * 16 + mn;
    const float bb = bias[co];
    #pragma unroll
    for (int t = 0; t < MT; ++t) {
        #pragma unroll
        for (int v = 0; v < 8; ++v)
            sOut[(t * 16 + 8 * (lane >> 4) + v) * 16 + mn] =
                (half_t)fmaxf(acc[t][v] + bb, 0.f);
    }
    __syncthreads();
    const int px = lane >> 1, h8 = (lane & 1) * 8;
    #pragma unroll
    for (int t = 0; t < MT; ++t) {
        v8h q = *(v8h*)&sOut[(t * 16 + px) * 16 + h8];
        *(v8h*)(out + ((((size_t)b * H + y0) * W) + x0 + 16 * t + px) * Cout
                + ct * 16 + h8) = q;
    }
}

// ---------------------------------------------------------------------------
__global__ void maxpool2(const half_t* __restrict__ in, half_t* __restrict__ out,
                         int C, int H, int W, int total)
{
    int idx = blockIdx.x * blockDim.x + threadIdx.x;
    if (idx >= total) return;
    int c = idx % C; int t = idx / C;
    int x = t % (W / 2); t /= (W / 2);
    int y = t % (H / 2); int b = t / (H / 2);
    size_t base = (((size_t)b * H + 2 * y) * W + 2 * x) * C + c;
    size_t rstep = (size_t)W * C;
    float v = fmaxf(fmaxf((float)in[base], (float)in[base + C]),
                    fmaxf((float)in[base + rstep], (float)in[base + rstep + C]));
    out[idx] = (half_t)v;
}

// ---------------------------------------------------------------------------
// Deterministic strided subset sample (NHWC: both src and dst contiguous).
// ---------------------------------------------------------------------------
__global__ void sample_feats(const half_t* __restrict__ feat,
                             half_t* __restrict__ xs, float* __restrict__ xn,
                             int C, int HW, int ns, int stride)
{
    int i = blockIdx.x * blockDim.x + threadIdx.x;
    int b = blockIdx.y;
    if (i >= ns) return;
    const half_t* src = feat + ((size_t)b * HW + (size_t)i * stride) * C;
    half_t* dst = xs + ((size_t)b * ns + i) * C;
    float ss = 0.f;
    for (int c = 0; c < C; c += 8) {
        v8h v = *(const v8h*)(src + c);
        #pragma unroll
        for (int q = 0; q < 8; ++q) { float f = (float)v[q]; ss += f * f; }
        *(v8h*)(dst + c) = v;
    }
    xn[b * ns + i] = sqrtf(ss);
}

__global__ void col_mean(const half_t* __restrict__ xs, float* __restrict__ mu,
                         int C, int ns, int total)
{
    int idx = blockIdx.x * blockDim.x + threadIdx.x;
    if (idx >= total) return;
    int c = idx % C; int b = idx / C;
    float s = 0.f;
    for (int i = 0; i < ns; ++i) s += (float)xs[((size_t)b * ns + i) * C + c];
    mu[idx] = s / (float)ns;
}

__global__ void center_transpose(const half_t* __restrict__ xs,
                                 const float* __restrict__ mu,
                                 half_t* __restrict__ xcT,
                                 int C, int ns, int total)
{
    int idx = blockIdx.x * blockDim.x + threadIdx.x;
    if (idx >= total) return;
    int i = idx % ns; int t = idx / ns;
    int c = t % C;    int b = t / C;
    xcT[idx] = (half_t)((float)xs[((size_t)b * ns + i) * C + c] - mu[b * C + c]);
}

// ---------------------------------------------------------------------------
// Cosine-distance tile (16 gen x 16 target) via WMMA; row/col minima reduced
// in-register (shfl within the lane groups that share an index per the wave32
// C/D layout) and committed with monotone-uint atomicMin. No NxN matrix.
// ---------------------------------------------------------------------------
__global__ __launch_bounds__(32) void pairdist_wmma(
    const half_t* __restrict__ xs, const float* __restrict__ xn,
    const half_t* __restrict__ ys, const float* __restrict__ yn,
    unsigned* __restrict__ minX, unsigned* __restrict__ minY,
    int C, int ns)
{
    const int lane = threadIdx.x;
    const int i0 = blockIdx.x * 16;
    const int j0 = blockIdx.y * 16;
    const int b  = blockIdx.z;
    const int mn  = lane & 15;
    const int hbA = (lane >> 4) * 8;
    const int hbB = (lane >> 4) * 8;

    const half_t* xrow = xs + ((size_t)b * ns + (i0 + mn)) * C;
    const half_t* yrow = ys + ((size_t)b * ns + (j0 + mn)) * C;

    v8f acc = {0.f, 0.f, 0.f, 0.f, 0.f, 0.f, 0.f, 0.f};
    for (int k = 0; k < C; k += 32) {
        HFrag a, bm;
        a.h[0]  = *(const v8h*)(xrow + k + hbA);
        a.h[1]  = *(const v8h*)(xrow + k + hbA + 16);
        bm.h[0] = *(const v8h*)(yrow + k + hbB);
        bm.h[1] = *(const v8h*)(yrow + k + hbB + 16);
        acc = __builtin_amdgcn_wmma_f32_16x16x32_f16(
            false, a.v, false, bm.v, (short)0, acc, false, false);
    }

    const int j = j0 + mn;
    const float ynj = yn[b * ns + j];
    float minOverI = 3.0e38f;
    #pragma unroll
    for (int v = 0; v < 8; ++v) {
        int i = i0 + (lane >> 4) * 8 + v;
        float d = 1.f - acc[v] / (xn[b * ns + i] * ynj + 1e-10f);
        minOverI = fminf(minOverI, d);
        float dm = d;
        for (int o = 1; o < 16; o <<= 1) dm = fminf(dm, __shfl_xor(dm, o, 32));
        if (mn == 0) atomicMin(&minY[b * ns + i], fenc(dm));
    }
    minOverI = fminf(minOverI, __shfl_xor(minOverI, 16, 32));
    if (lane < 16) atomicMin(&minX[b * ns + j], fenc(minOverI));
}

__global__ void style_finalize(const unsigned* __restrict__ minX,
                               const unsigned* __restrict__ minY,
                               int ns, float* __restrict__ loss)
{
    int b = blockIdx.x, tid = threadIdx.x;
    __shared__ float sx[256], sy[256];
    float ax = 0.f, ay = 0.f;
    for (int i = tid; i < ns; i += 256) {
        ax += fdec(minX[b * ns + i]);
        ay += fdec(minY[b * ns + i]);
    }
    sx[tid] = ax; sy[tid] = ay;
    __syncthreads();
    for (int o = 128; o > 0; o >>= 1) {
        if (tid < o) { sx[tid] += sx[tid + o]; sy[tid] += sy[tid + o]; }
        __syncthreads();
    }
    if (tid == 0) atomicAdd(&loss[b], fmaxf(sx[0], sy[0]) / (float)ns);
}

// ---------------------------------------------------------------------------
// Fused covariance-difference: two WMMA accumulators (x and y covariance
// tiles, K = ns), |diff| reduced in-register; covariances never materialized.
// ---------------------------------------------------------------------------
__global__ __launch_bounds__(32) void covdiff_wmma(
    const half_t* __restrict__ xcT, const half_t* __restrict__ ycT,
    float* __restrict__ loss, int C, int ns)
{
    const int lane = threadIdx.x;
    const int c0 = blockIdx.x * 16;
    const int d0 = blockIdx.y * 16;
    const int b  = blockIdx.z;
    const int mn  = lane & 15;
    const int hbA = (lane >> 4) * 8;
    const int hbB = (lane >> 4) * 8;

    const half_t* xa = xcT + ((size_t)b * C + (c0 + mn)) * ns;
    const half_t* xb = xcT + ((size_t)b * C + (d0 + mn)) * ns;
    const half_t* ya = ycT + ((size_t)b * C + (c0 + mn)) * ns;
    const half_t* yb = ycT + ((size_t)b * C + (d0 + mn)) * ns;

    v8f ax = {0.f, 0.f, 0.f, 0.f, 0.f, 0.f, 0.f, 0.f};
    v8f ay = {0.f, 0.f, 0.f, 0.f, 0.f, 0.f, 0.f, 0.f};
    for (int k = 0; k < ns; k += 32) {
        HFrag a, bm;
        a.h[0]  = *(const v8h*)(xa + k + hbA);
        a.h[1]  = *(const v8h*)(xa + k + hbA + 16);
        bm.h[0] = *(const v8h*)(xb + k + hbB);
        bm.h[1] = *(const v8h*)(xb + k + hbB + 16);
        ax = __builtin_amdgcn_wmma_f32_16x16x32_f16(
            false, a.v, false, bm.v, (short)0, ax, false, false);
        a.h[0]  = *(const v8h*)(ya + k + hbA);
        a.h[1]  = *(const v8h*)(ya + k + hbA + 16);
        bm.h[0] = *(const v8h*)(yb + k + hbB);
        bm.h[1] = *(const v8h*)(yb + k + hbB + 16);
        ay = __builtin_amdgcn_wmma_f32_16x16x32_f16(
            false, a.v, false, bm.v, (short)0, ay, false, false);
    }

    const float inv = 1.f / (float)(ns - 1);
    float s = 0.f;
    #pragma unroll
    for (int v = 0; v < 8; ++v) s += fabsf(ax[v] * inv - ay[v] * inv);
    for (int o = 1; o < 32; o <<= 1) s += __shfl_xor(s, o, 32);
    if (lane == 0) atomicAdd(&loss[b], s / ((float)C * (float)C));
}

__global__ void mu_diff(const float* __restrict__ xmu, const float* __restrict__ ymu,
                        int C, float* __restrict__ loss)
{
    int b = blockIdx.x, tid = threadIdx.x;
    __shared__ float s[128];
    float a = 0.f;
    for (int c = tid; c < C; c += 128) a += fabsf(xmu[b * C + c] - ymu[b * C + c]);
    s[tid] = a;
    __syncthreads();
    for (int o = 64; o > 0; o >>= 1) {
        if (tid < o) s[tid] += s[tid + o];
        __syncthreads();
    }
    if (tid == 0) atomicAdd(&loss[b], s[0] / (float)C);
}

// ---------------------------------------------------------------------------
extern "C" void kernel_launch(void* const* d_in, const int* in_sizes, int n_in,
                              void* d_out, int out_size, void* d_ws, size_t ws_size,
                              hipStream_t stream)
{
    (void)in_sizes; (void)n_in; (void)ws_size;
    const int B = 4;
    const float* imgT = (const float*)d_in[0];
    const float* imgG = (const float*)d_in[1];
    const float* Wt[11]; const float* Bi[11];
    for (int i = 0; i < 11; ++i) {
        Wt[i] = (const float*)d_in[2 + 2 * i];
        Bi[i] = (const float*)d_in[3 + 2 * i];
    }
    float* loss = (float*)d_out;

    char* ws = (char*)d_ws;
    size_t off = 0;
    auto alloc = [&](size_t bytes) -> char* {
        size_t r = off;
        off += (bytes + 255) & ~(size_t)255;
        return ws + r;
    };
    const size_t ACT = (size_t)B * 64 * 256 * 256 * 2;
    half_t* bufA = (half_t*)alloc(ACT);
    half_t* bufB = (half_t*)alloc(ACT);
    const size_t SH = (size_t)B * 1024 * 512 * 2;
    half_t* ysH[5]; half_t* ycT[5]; float* ynm[5]; float* ymu[5];
    for (int l = 0; l < 5; ++l) {
        ysH[l] = (half_t*)alloc(SH);
        ycT[l] = (half_t*)alloc(SH);
        ynm[l] = (float*)alloc((size_t)B * 1024 * 4);
        ymu[l] = (float*)alloc((size_t)B * 512 * 4);
    }
    half_t* xsH = (half_t*)alloc(SH);
    half_t* xcTg = (half_t*)alloc(SH);
    float* xnm = (float*)alloc((size_t)B * 1024 * 4);
    float* xmu = (float*)alloc((size_t)B * 512 * 4);
    unsigned* minX = (unsigned*)alloc((size_t)B * 1024 * 4);
    unsigned* minY = (unsigned*)alloc((size_t)B * 1024 * 4);

    // packed f16 weights (B-fragment order), once per call
    static const int CC[11][2] = {{3,64},{64,64},{64,128},{128,128},{128,256},
                                  {256,256},{256,256},{256,512},{512,512},
                                  {512,512},{512,512}};
    half_t* wpk[11];
    for (int i = 0; i < 11; ++i) {
        int nch = (CC[i][0] + 31) / 32;
        int total = (CC[i][1] / 16) * nch * 9 * 16 * 32;
        wpk[i] = (half_t*)alloc((size_t)total * 2);
        pack_weights<<<dim3((total + 255) / 256), dim3(256), 0, stream>>>(
            Wt[i], wpk[i], CC[i][0], nch, total);
    }

    (void)hipMemsetAsync(d_out, 0, sizeof(float) * out_size, stream);

    auto conv = [&](const half_t* inHp, const float* inFp, int ci,
                    int Cin, int Cout, int H, int Wd, half_t* o) {
        int mt = (Wd % 32 == 0) ? 2 : 1;
        dim3 g(Wd / (16 * mt), H, B * (Cout / 16));
        if (mt == 2)
            conv3x3_wmma<2><<<g, dim3(32), 0, stream>>>(inHp, inFp, wpk[ci], Bi[ci],
                                                        o, Cin, Cout, H, Wd);
        else
            conv3x3_wmma<1><<<g, dim3(32), 0, stream>>>(inHp, inFp, wpk[ci], Bi[ci],
                                                        o, Cin, Cout, H, Wd);
    };
    auto pool = [&](const half_t* in, int C, int H, int Wd, half_t* o) {
        int total = B * C * (H / 2) * (Wd / 2);
        maxpool2<<<dim3((total + 255) / 256), dim3(256), 0, stream>>>(in, o, C, H, Wd, total);
    };
    auto tap = [&](int phase, int l, const half_t* feat, int C, int HW) {
        int ns = HW < 1024 ? HW : 1024;
        int stride = HW / ns;
        half_t* sH = phase ? xsH  : ysH[l];
        half_t* cT = phase ? xcTg : ycT[l];
        float*  nm = phase ? xnm  : ynm[l];
        float*  mu = phase ? xmu  : ymu[l];
        sample_feats<<<dim3((ns + 63) / 64, B), dim3(64), 0, stream>>>(feat, sH, nm, C, HW, ns, stride);
        col_mean<<<dim3((B * C + 127) / 128), dim3(128), 0, stream>>>(sH, mu, C, ns, B * C);
        center_transpose<<<dim3((B * C * ns + 255) / 256), dim3(256), 0, stream>>>(sH, mu, cT, C, ns, B * C * ns);
        if (phase) {
            (void)hipMemsetAsync(minX, 0xFF, (size_t)B * ns * 4, stream);
            (void)hipMemsetAsync(minY, 0xFF, (size_t)B * ns * 4, stream);
            pairdist_wmma<<<dim3(ns / 16, ns / 16, B), dim3(32), 0, stream>>>(
                xsH, xnm, ysH[l], ynm[l], minX, minY, C, ns);
            style_finalize<<<dim3(B), dim3(256), 0, stream>>>(minX, minY, ns, loss);
            covdiff_wmma<<<dim3(C / 16, C / 16, B), dim3(32), 0, stream>>>(xcTg, ycT[l], loss, C, ns);
            mu_diff<<<dim3(B), dim3(128), 0, stream>>>(xmu, ymu[l], C, loss);
        }
    };

    for (int phase = 0; phase < 2; ++phase) {
        const float* img = phase ? imgG : imgT;
        conv(nullptr, img, 0, 3, 64, 256, 256, bufA);          // conv1_1 + relu
        tap(phase, 0, bufA, 64, 256 * 256);                    // style layer 1
        conv(bufA, nullptr, 1, 64, 64, 256, 256, bufB);        // conv1_2
        pool(bufB, 64, 256, 256, bufA);                        // pool1
        conv(bufA, nullptr, 2, 64, 128, 128, 128, bufB);       // conv2_1
        tap(phase, 1, bufB, 128, 128 * 128);                   // style layer 6
        conv(bufB, nullptr, 3, 128, 128, 128, 128, bufA);      // conv2_2
        pool(bufA, 128, 128, 128, bufB);                       // pool2
        conv(bufB, nullptr, 4, 128, 256, 64, 64, bufA);        // conv3_1
        tap(phase, 2, bufA, 256, 64 * 64);                     // style layer 11
        conv(bufA, nullptr, 5, 256, 256, 64, 64, bufB);        // conv3_2
        conv(bufB, nullptr, 6, 256, 256, 64, 64, bufA);        // conv3_3
        pool(bufA, 256, 64, 64, bufB);                         // pool3
        conv(bufB, nullptr, 7, 256, 512, 32, 32, bufA);        // conv4_1
        tap(phase, 3, bufA, 512, 32 * 32);                     // style layer 18
        conv(bufA, nullptr, 8, 512, 512, 32, 32, bufB);        // conv4_2
        conv(bufB, nullptr, 9, 512, 512, 32, 32, bufA);        // conv4_3
        pool(bufA, 512, 32, 32, bufB);                         // pool4
        conv(bufB, nullptr, 10, 512, 512, 16, 16, bufA);       // conv5_1
        tap(phase, 4, bufA, 512, 16 * 16);                     // style layer 25
    }
}
